// DeformableTransformerLayer_27109833572611
// MI455X (gfx1250) — matrix operand order repaired and checked by reading, compiled, not run on hardware
//
#include <hip/hip_runtime.h>

// ---------------------------------------------------------------------------
// DeformableTransformerLayer for MI455X (gfx1250, wave32, WMMA 16x16x32 f16)
// ---------------------------------------------------------------------------

typedef __attribute__((ext_vector_type(16))) _Float16 hvec16;
typedef __attribute__((ext_vector_type(8)))  _Float16 hvec8;
typedef __attribute__((ext_vector_type(2)))  _Float16 hvec2;
typedef __attribute__((ext_vector_type(8)))  float    fvec8;

namespace cfg {
constexpr int B  = 2;
constexpr int Q  = 16384;      // 128*128
constexpr int C  = 256;
constexpr int H  = 8;
constexpr int D  = 32;         // C / H
constexpr int L  = 3;
constexpr int P  = 4;
constexpr int F  = 1024;       // C * MLP
constexpr int NV = 13125;      // 100*100 + 50*50 + 25*25
constexpr int HL[3] = {100, 50, 25};
constexpr int WL[3] = {100, 50, 25};
constexpr int LS[3] = {0, 10000, 12500};
}

// ---------------------------------------------------------------------------
// LayerNorm (wave per row of 256), optional +query_pos, f16 output
// ---------------------------------------------------------------------------
template<bool ADDPOS>
__global__ __launch_bounds__(256)
void ln_kernel(const float* __restrict__ x, const float* __restrict__ g,
               const float* __restrict__ bb, const float* __restrict__ pos,
               _Float16* __restrict__ out, int rows)
{
    int row  = blockIdx.x * 8 + (threadIdx.x >> 5);
    int lane = threadIdx.x & 31;
    if (row >= rows) return;

    const float* xr = x + (size_t)row * cfg::C;
    float4 v0 = *(const float4*)(xr + lane * 8);
    float4 v1 = *(const float4*)(xr + lane * 8 + 4);
    float vals[8] = {v0.x, v0.y, v0.z, v0.w, v1.x, v1.y, v1.z, v1.w};

    float s = 0.f, ss = 0.f;
#pragma unroll
    for (int j = 0; j < 8; ++j) { s += vals[j]; ss += vals[j] * vals[j]; }
#pragma unroll
    for (int m = 16; m >= 1; m >>= 1) {
        s  += __shfl_xor(s,  m, 32);
        ss += __shfl_xor(ss, m, 32);
    }
    float mean = s * (1.f / cfg::C);
    float var  = ss * (1.f / cfg::C) - mean * mean;
    float rs   = rsqrtf(var + 1e-5f);

#pragma unroll
    for (int j = 0; j < 8; ++j) {
        int c = lane * 8 + j;
        float y = (vals[j] - mean) * rs * g[c] + bb[c];
        if (ADDPOS) y += pos[(size_t)row * cfg::C + c];
        out[(size_t)row * cfg::C + c] = (_Float16)y;
    }
}

// ---------------------------------------------------------------------------
// f32 -> f16 elementwise
// ---------------------------------------------------------------------------
__global__ __launch_bounds__(256)
void cvt_f32_f16(const float* __restrict__ in, _Float16* __restrict__ out, int n)
{
    int i = blockIdx.x * 256 + threadIdx.x;
    if (i < n) out[i] = (_Float16)in[i];
}

// ---------------------------------------------------------------------------
// Weight pre-pass: f32 KxN row-major  ->  f16 NxK (transposed)
// Tiny (<=256K elements per weight), runs once per launch.
// ---------------------------------------------------------------------------
__global__ __launch_bounds__(256)
void wtx_kernel(const float* __restrict__ W, _Float16* __restrict__ out,
                int K, int N)
{
    int i = blockIdx.x * 256 + threadIdx.x;   // output index: n*K + k
    if (i >= K * N) return;
    int n = i / K;
    int k = i - n * K;
    out[i] = (_Float16)W[(size_t)k * N + n];
}

// ---------------------------------------------------------------------------
// WMMA GEMM: out = epilogue(A[f16, MxK] * Wt[f16, NxK pre-transposed] + bias)
// Block: 256 thr (8 waves in 4x2), tile 128x64, wave computes 32x32
// (four 16x16x32 WMMAs per K-step, A frags reused over N, B frags over M).
// EPI: 0 = f32 store, 2 = f32 store + residual, 3 = GELU + f16 store
// ---------------------------------------------------------------------------
template<int EPI>
__global__ __launch_bounds__(256)
void wmma_gemm(const _Float16* __restrict__ A, const _Float16* __restrict__ Wt,
               const float* __restrict__ bias, const float* __restrict__ res,
               void* __restrict__ outp, int M, int N, int K)
{
    __shared__ __align__(16) _Float16 As[128][40];  // [m][k]  (+pad)
    __shared__ __align__(16) _Float16 Bs[64][40];   // [n][k]  (+pad)

    const int t    = threadIdx.x;
    const int lane = t & 31, wave = t >> 5;
    const int wm   = wave >> 1, wn = wave & 1;      // 4x2 wave grid
    const int h    = lane >> 4, ml = lane & 15;     // lane group / lane-in-16
    const int m0   = blockIdx.y * 128;
    const int n0   = blockIdx.x * 64;

    fvec8 acc[2][2] = {{{0,0,0,0,0,0,0,0}, {0,0,0,0,0,0,0,0}},
                       {{0,0,0,0,0,0,0,0}, {0,0,0,0,0,0,0,0}}};

    const int ar  = t >> 1;          // A staging: row 0..127
    const int ak  = (t & 1) * 16;    //            k  0 / 16
    const int bnn = t & 63;          // B staging: n 0..63 (one row of Wt)
    const int bkc = (t >> 6) * 8;    //            k chunk 0/8/16/24

    const bool ain = (m0 + ar)  < M;
    const bool bin = (n0 + bnn) < N; // N is always a multiple of 8 here

    const _Float16* ap = A  + (size_t)(m0 + ar)  * K + ak;
    const _Float16* bp = Wt + (size_t)(n0 + bnn) * K + bkc;

    for (int k0 = 0; k0 < K; k0 += 32) {
        // ---- stage A slab (128x32 f16): two b128 loads, two b128 stores ----
        uint4 pa0 = make_uint4(0u, 0u, 0u, 0u);
        uint4 pa1 = make_uint4(0u, 0u, 0u, 0u);
        if (ain) {
            pa0 = *(const uint4*)(ap + k0);
            pa1 = *(const uint4*)(ap + k0 + 8);
            if (k0 + 32 < K)
                __builtin_prefetch(ap + k0 + 32, 0, 1);
        }
        *(uint4*)&As[ar][ak]     = pa0;
        *(uint4*)&As[ar][ak + 8] = pa1;

        // ---- stage B slab (64x32 f16 from pre-transposed weights) ----
        uint4 pb = make_uint4(0u, 0u, 0u, 0u);
        if (bin)
            pb = *(const uint4*)(bp + k0);
        *(uint4*)&Bs[bnn][bkc] = pb;
        __syncthreads();

        // ---- A fragments (ISA 7.12.2: lanes->M, regs->interleaved K) ----
        // per lane group h the 16 elements form two contiguous 8-half runs
        hvec16 af[2];
#pragma unroll
        for (int mt = 0; mt < 2; ++mt) {
            const int rl = wm * 32 + mt * 16 + ml;
#pragma unroll
            for (int p = 0; p < 8; ++p) {
                int kk = ((p >= 4) ? 16 : 0) + (p & 3) * 2 + 8 * h;
                hvec2 pr = *(const hvec2*)&As[rl][kk];
                af[mt][2 * p]     = pr[0];
                af[mt][2 * p + 1] = pr[1];
            }
        }

        // ---- B fragments (lanes->N, regs->interleaved K, same as A) ----
        hvec16 bf[2];
#pragma unroll
        for (int nt = 0; nt < 2; ++nt) {
            const int nl = wn * 32 + nt * 16 + ml;
            hvec8 blo = *(const hvec8*)&Bs[nl][8 * h];
            hvec8 bhi = *(const hvec8*)&Bs[nl][16 + 8 * h];
            bf[nt] = __builtin_shufflevector(blo, bhi,
                         0,1,2,3,4,5,6,7,8,9,10,11,12,13,14,15);
        }

#pragma unroll
        for (int mt = 0; mt < 2; ++mt)
#pragma unroll
            for (int nt = 0; nt < 2; ++nt)
                acc[mt][nt] = __builtin_amdgcn_wmma_f32_16x16x32_f16(
                    false, af[mt], false, bf[nt], (short)0, acc[mt][nt],
                    false, false);
        __syncthreads();
    }

    // ---- epilogue (D layout: m = r + 8*h, n = lane&15) ----
#pragma unroll
    for (int mt = 0; mt < 2; ++mt) {
#pragma unroll
        for (int r = 0; r < 8; ++r) {
            int row = m0 + wm * 32 + mt * 16 + 8 * h + r;
            if (row >= M) continue;
#pragma unroll
            for (int nt = 0; nt < 2; ++nt) {
                int col = n0 + wn * 32 + nt * 16 + ml;
                if (col >= N) continue;
                float x = acc[mt][nt][r] + bias[col];
                size_t oi = (size_t)row * N + col;
                if constexpr (EPI == 0) {
                    ((float*)outp)[oi] = x;
                } else if constexpr (EPI == 2) {
                    ((float*)outp)[oi] = x + res[oi];
                } else if constexpr (EPI == 3) {
                    float gx = 0.5f * x * (1.0f + erff(x * 0.70710678118654752f));
                    ((_Float16*)outp)[oi] = (_Float16)gx;
                }
            }
        }
    }
}

// ---------------------------------------------------------------------------
// Softmax over 12 contiguous (L*P) attention logits, in place
// ---------------------------------------------------------------------------
__global__ __launch_bounds__(256)
void softmax12(float* __restrict__ a, int groups)
{
    int gi = blockIdx.x * 256 + threadIdx.x;
    if (gi >= groups) return;
    float* p = a + (size_t)gi * 12;
    float mx = p[0];
#pragma unroll
    for (int i = 1; i < 12; ++i) mx = fmaxf(mx, p[i]);
    float e[12], sum = 0.f;
#pragma unroll
    for (int i = 0; i < 12; ++i) { e[i] = __expf(p[i] - mx); sum += e[i]; }
    float inv = 1.f / sum;
#pragma unroll
    for (int i = 0; i < 12; ++i) p[i] = e[i] * inv;
}

// ---------------------------------------------------------------------------
// Deformable bilinear sampling: wave per (b,q,head), lane = channel (d=32)
// grid_sample(align_corners=False, padding=zeros) semantics
// ---------------------------------------------------------------------------
__global__ __launch_bounds__(256)
void sample_kernel(const float* __restrict__ v, const float* __restrict__ ref,
                   const float* __restrict__ off, const float* __restrict__ aw,
                   _Float16* __restrict__ out)
{
    int wid  = blockIdx.x * 8 + (threadIdx.x >> 5);   // (b*Q+q)*H + hd
    int lane = threadIdx.x & 31;                      // channel
    int hd   = wid & 7;
    int bq   = wid >> 3;
    int b    = bq >> 14;                              // Q = 2^14

    float rx = ref[bq * 2 + 0];
    float ry = ref[bq * 2 + 1];
    const float* offp = off + (size_t)wid * 24;
    const float* awp  = aw  + (size_t)wid * 12;

    float acc = 0.f;
#pragma unroll
    for (int l = 0; l < cfg::L; ++l) {
        const int Hl = cfg::HL[l], Wl = cfg::WL[l];
        // value stride per spatial pos = H*D = 256 floats
        const size_t vbase =
            (((size_t)b * cfg::NV + cfg::LS[l]) * cfg::H + hd) * cfg::D + lane;
#pragma unroll
        for (int p = 0; p < cfg::P; ++p) {
            int   lp = l * cfg::P + p;
            float ox = offp[lp * 2 + 0];
            float oy = offp[lp * 2 + 1];
            float a  = awp[lp];
            // loc = ref + off/(W,H);  pixel = loc*size - 0.5
            float x = rx * (float)Wl + ox - 0.5f;
            float y = ry * (float)Hl + oy - 0.5f;
            float x0f = floorf(x), y0f = floorf(y);
            float lx = x - x0f, ly = y - y0f;
            int   x0 = (int)x0f, y0 = (int)y0f;

            auto fetch = [&](int xi, int yi) -> float {
                bool valid = (xi >= 0) && (xi < Wl) && (yi >= 0) && (yi < Hl);
                int xc = min(max(xi, 0), Wl - 1);
                int yc = min(max(yi, 0), Hl - 1);
                float g = v[vbase + (size_t)(yc * Wl + xc) * (cfg::H * cfg::D)];
                return valid ? g : 0.f;
            };
            float s = fetch(x0,     y0    ) * (1.f - lx) * (1.f - ly)
                    + fetch(x0 + 1, y0    ) * lx         * (1.f - ly)
                    + fetch(x0,     y0 + 1) * (1.f - lx) * ly
                    + fetch(x0 + 1, y0 + 1) * lx         * ly;
            acc += a * s;
        }
    }
    out[(size_t)wid * cfg::D + lane] = (_Float16)acc;
}

// ---------------------------------------------------------------------------
// Launch
// ---------------------------------------------------------------------------
extern "C" void kernel_launch(void* const* d_in, const int* in_sizes, int n_in,
                              void* d_out, int out_size, void* d_ws, size_t ws_size,
                              hipStream_t stream)
{
    (void)in_sizes; (void)n_in; (void)out_size; (void)ws_size;

    const float* query  = (const float*)d_in[0];
    const float* value  = (const float*)d_in[1];
    const float* qpos   = (const float*)d_in[2];
    const float* ref    = (const float*)d_in[3];
    const float* ln1_g  = (const float*)d_in[6];
    const float* ln1_b  = (const float*)d_in[7];
    const float* ln2_g  = (const float*)d_in[8];
    const float* ln2_b  = (const float*)d_in[9];
    const float* W_value= (const float*)d_in[10];
    const float* b_value= (const float*)d_in[11];
    const float* W_off  = (const float*)d_in[12];
    const float* b_off  = (const float*)d_in[13];
    const float* W_attn = (const float*)d_in[14];
    const float* b_attn = (const float*)d_in[15];
    const float* W_out  = (const float*)d_in[16];
    const float* b_out  = (const float*)d_in[17];
    const float* W_ffn1 = (const float*)d_in[18];
    const float* b_ffn1 = (const float*)d_in[19];
    const float* W_ffn2 = (const float*)d_in[20];
    const float* b_ffn2 = (const float*)d_in[21];
    float* out = (float*)d_out;

    // ---- workspace layout (with lifetime-based aliasing) ----
    char* ws = (char*)d_ws;
    _Float16* q16 = (_Float16*)ws;                       // 16 MiB (q16, later h16)
    char* r1 = ws + (size_t)16 * 1024 * 1024;
    float*    v32    = (float*)r1;                                        // 26,880,000 B
    float*    off32  = (float*)(r1 + 26880000);                           // 25,165,824 B
    float*    attn32 = (float*)(r1 + 26880000 + 25165824);                // 12,582,912 B
    _Float16* samp16 = (_Float16*)(r1 + 26880000 + 25165824 + 12582912);  // 16,777,216 B
    _Float16* val16  = samp16;          // value-f16 dead before sampling writes
    _Float16* g16    = (_Float16*)r1;   // FFN1 out (64 MiB) reuses v/off/attn/samp
    _Float16* h16    = q16;             // LN2 out reuses q16

    // persistent pre-transposed f16 weights (after region1's max extent)
    char* wtb = r1 + 81405952;
    _Float16* Wv16  = (_Float16*)wtb;              // 256*256
    _Float16* Wo16  = Wv16  + 256 * 256;           // 192*256 (NxK)
    _Float16* Wa16  = Wo16  + 192 * 256;           // 96*256
    _Float16* Wu16  = Wa16  +  96 * 256;           // 256*256
    _Float16* Wf1t  = Wu16  + 256 * 256;           // 1024*256
    _Float16* Wf2t  = Wf1t  + 1024 * 256;          // 256*1024

    const int rowsQ = cfg::B * cfg::Q;   // 32768
    const int rowsV = cfg::B * cfg::NV;  // 26250
    const int groups = rowsQ * cfg::H;   // 262144

    // 0. weight pre-pass: cvt+transpose to f16 NxK
    wtx_kernel<<<(256 * 256 + 255) / 256, 256, 0, stream>>>(W_value, Wv16, 256, 256);
    wtx_kernel<<<(256 * 192 + 255) / 256, 256, 0, stream>>>(W_off,   Wo16, 256, 192);
    wtx_kernel<<<(256 *  96 + 255) / 256, 256, 0, stream>>>(W_attn,  Wa16, 256,  96);
    wtx_kernel<<<(256 * 256 + 255) / 256, 256, 0, stream>>>(W_out,   Wu16, 256, 256);
    wtx_kernel<<<(256 * 1024 + 255) / 256, 256, 0, stream>>>(W_ffn1, Wf1t, 256, 1024);
    wtx_kernel<<<(1024 * 256 + 255) / 256, 256, 0, stream>>>(W_ffn2, Wf2t, 1024, 256);

    // 1. q = LN1(query) + query_pos  -> f16
    ln_kernel<true><<<rowsQ / 8, 256, 0, stream>>>(query, ln1_g, ln1_b, qpos, q16, rowsQ);

    // 2. value -> f16
    int nvE = rowsV * cfg::C;
    cvt_f32_f16<<<(nvE + 255) / 256, 256, 0, stream>>>(value, val16, nvE);

    // 3. value projection (f32 out, kept for sampling accuracy)
    wmma_gemm<0><<<dim3(4, (rowsV + 127) / 128), 256, 0, stream>>>(
        val16, Wv16, b_value, nullptr, v32, rowsV, 256, 256);

    // 4/5. sampling offsets + attention logits
    wmma_gemm<0><<<dim3(3, rowsQ / 128), 256, 0, stream>>>(
        q16, Wo16, b_off, nullptr, off32, rowsQ, 192, 256);
    wmma_gemm<0><<<dim3(2, rowsQ / 128), 256, 0, stream>>>(
        q16, Wa16, b_attn, nullptr, attn32, rowsQ, 96, 256);

    // 6. softmax over L*P
    softmax12<<<groups / 256, 256, 0, stream>>>(attn32, groups);

    // 7. deformable bilinear sampling -> f16
    sample_kernel<<<groups / 8, 256, 0, stream>>>(v32, ref, off32, attn32, samp16);

    // 8. x = query + sampled @ W_out + b_out   -> d_out
    wmma_gemm<2><<<dim3(4, rowsQ / 128), 256, 0, stream>>>(
        samp16, Wu16, b_out, query, out, rowsQ, 256, 256);

    // 9. h = LN2(x) -> f16
    ln_kernel<false><<<rowsQ / 8, 256, 0, stream>>>(out, ln2_g, ln2_b, nullptr, h16, rowsQ);

    // 10. FFN1 + GELU -> f16
    wmma_gemm<3><<<dim3(16, rowsQ / 128), 256, 0, stream>>>(
        h16, Wf1t, b_ffn1, nullptr, g16, rowsQ, 1024, 256);

    // 11. out = x + FFN2(g)   (in-place residual on d_out)
    wmma_gemm<2><<<dim3(4, rowsQ / 128), 256, 0, stream>>>(
        g16, Wf2t, b_ffn2, out, out, rowsQ, 256, 1024);
}
